// GCNConvDGL_11682311045155
// MI455X (gfx1250) — matrix operand order repaired and checked by reading, not compiled, not run on hardware
//
#include <hip/hip_runtime.h>
#include <hip/hip_bf16.h>

// ---------------------------------------------------------------------------
// GCN GraphConv (norm='none'):  out = segment_sum(x[src], dst) @ W + b
// Phase 1: zero agg (reuses d_out as the f32 agg buffer)
// Phase 2: split+permute W into bf16 hi/lo WMMA B-fragments (d_ws)
// Phase 3: edge-parallel gather + f32 atomic scatter-add into agg
// Phase 4: WMMA GEMM, bf16 hi/lo split (3 wmma per K-step, f32 accum),
//          in-place agg -> out with a workgroup barrier between read & write
// ---------------------------------------------------------------------------

#define N_NODES 50000
#define D_FEAT  512            // D_IN == D_OUT == 512

typedef __attribute__((ext_vector_type(16))) __bf16 v16bf;
typedef __attribute__((ext_vector_type(8)))  float  v8f;

union BfPack {
    v16bf v;
    unsigned short u[16];
};

__device__ __forceinline__ unsigned short f2bf(float f) {
    // round-to-nearest-even f32 -> bf16 (NaN payloads don't matter here)
    unsigned int u = __float_as_uint(f);
    u += 0x7fffu + ((u >> 16) & 1u);
    return (unsigned short)(u >> 16);
}

__device__ __forceinline__ float bf2f(unsigned short h) {
    return __uint_as_float(((unsigned int)h) << 16);
}

__device__ __forceinline__ v8f wmma_bf16(v16bf a, v16bf b, v8f c) {
    // D = A(16x32 bf16) x B(32x16 bf16) + C(16x16 f32)
    return __builtin_amdgcn_wmma_f32_16x16x32_bf16(
        /*neg_a=*/false, a, /*neg_b=*/false, b,
        /*c_mod=*/(short)0, c, /*reuse_a=*/false, /*reuse_b=*/false);
}

// -------------------------------- Phase 1 ----------------------------------
__global__ __launch_bounds__(256) void zero_kernel(float4* agg4, int n4) {
    int i = blockIdx.x * blockDim.x + threadIdx.x;
    int stride = gridDim.x * blockDim.x;
    float4 z = make_float4(0.f, 0.f, 0.f, 0.f);
    for (; i < n4; i += stride) agg4[i] = z;
}

// -------------------------------- Phase 2 ----------------------------------
// Permuted layout: for K-tile kt (32 rows) and output column n, the 32 bf16
// values W[kt*32 .. kt*32+31][n] are stored contiguously at
//   perm[((kt*512 + n)*32) + (k % 32)]
// so lane (n=l16, half) loads one contiguous v16bf (32 B) per fragment.
__global__ __launch_bounds__(256) void wsplit_kernel(const float* __restrict__ weight,
                                                     unsigned short* __restrict__ w_hi,
                                                     unsigned short* __restrict__ w_lo) {
    int idx = blockIdx.x * blockDim.x + threadIdx.x;   // 0 .. 512*512-1
    int k = idx >> 9;          // input-feature row of W [512][512]
    int n = idx & 511;         // output column
    float w = weight[idx];
    unsigned short h = f2bf(w);
    unsigned short l = f2bf(w - bf2f(h));
    size_t o = ((size_t)((k >> 5) * D_FEAT + n) << 5) + (k & 31);
    w_hi[o] = h;
    w_lo[o] = l;
}

// -------------------------------- Phase 3 ----------------------------------
// One wave32 per edge; lane handles 16 floats as 4 float4 strided by 128 floats.
__global__ __launch_bounds__(256) void scatter_kernel(const float* __restrict__ x,
                                                      const int* __restrict__ src,
                                                      const int* __restrict__ dst,
                                                      float* __restrict__ agg,
                                                      int n_edges) {
    int edge = blockIdx.x * (blockDim.x >> 5) + (threadIdx.x >> 5);
    if (edge >= n_edges) return;                       // wave-uniform
    int lane = threadIdx.x & 31;
    int s = src[edge];
    int d = dst[edge];
    const float* xr = x + (size_t)s * D_FEAT + lane * 4;
    float* ar = agg + (size_t)d * D_FEAT + lane * 4;
#pragma unroll
    for (int j = 0; j < 4; ++j) {
        float4 v = *(const float4*)(xr + j * 128);
        atomicAdd(ar + j * 128 + 0, v.x);
        atomicAdd(ar + j * 128 + 1, v.y);
        atomicAdd(ar + j * 128 + 2, v.z);
        atomicAdd(ar + j * 128 + 3, v.w);
    }
}

// -------------------------------- Phase 4 ----------------------------------
// Block = 8 waves; block owns output rows [16*bx, 16*bx+16); wave w owns
// columns [64*w, 64*w+64) as 4 N-subtiles. K loop: 16 steps of 32.
// 3 bf16 WMMAs per (K-step, subtile): hi*hi + lo*hi + hi*lo  (~f32 precision).
__global__ __launch_bounds__(256) void gemm_kernel(const float* __restrict__ agg,
                                                   const unsigned short* __restrict__ w_hi,
                                                   const unsigned short* __restrict__ w_lo,
                                                   const float* __restrict__ bias,
                                                   float* __restrict__ out) {
    const int wave  = threadIdx.x >> 5;    // 0..7 -> 64-col slab
    const int lane  = threadIdx.x & 31;
    const int half  = lane >> 4;           // K sub-half select (A & B layouts)
    const int l16   = lane & 15;           // A: row M; B/C: column N
    const int mbase = blockIdx.x << 4;
    const int nbase = wave << 6;

    v8f c[4] = {};

    // A fragment source: row (mbase+l16); lane needs K = kb+half*16+[0,16)
    // (contiguous, matching the B fragment), i.e. one 64 B run.
    const float* arow = agg + (size_t)(mbase + l16) * D_FEAT + half * 16;

    for (int ks = 0; ks < 16; ++ks) {
        const float* p = arow + ks * 32;
        if (ks < 15) __builtin_prefetch(p + 32, 0, 1);   // global_prefetch_b8
        float4 f0 = *(const float4*)(p);
        float4 f1 = *(const float4*)(p + 4);
        float4 f2 = *(const float4*)(p + 8);
        float4 f3 = *(const float4*)(p + 12);

        float av[16];
        av[0]=f0.x;  av[1]=f0.y;  av[2]=f0.z;  av[3]=f0.w;
        av[4]=f1.x;  av[5]=f1.y;  av[6]=f1.z;  av[7]=f1.w;
        av[8]=f2.x;  av[9]=f2.y;  av[10]=f2.z; av[11]=f2.w;
        av[12]=f3.x; av[13]=f3.y; av[14]=f3.z; av[15]=f3.w;

        BfPack ah, al;
#pragma unroll
        for (int j = 0; j < 16; ++j) {
            unsigned short h = f2bf(av[j]);
            ah.u[j] = h;
            al.u[j] = f2bf(av[j] - bf2f(h));
        }

#pragma unroll
        for (int nsub = 0; nsub < 4; ++nsub) {
            int n = nbase + (nsub << 4) + l16;
            size_t bo = (((size_t)(ks * D_FEAT + n)) << 5) + half * 16;
            v16bf bh = *(const v16bf*)(w_hi + bo);   // 32 B contiguous
            v16bf bl = *(const v16bf*)(w_lo + bo);
            c[nsub] = wmma_bf16(ah.v, bh, c[nsub]);  // hi*hi
            c[nsub] = wmma_bf16(al.v, bh, c[nsub]);  // lo*hi
            c[nsub] = wmma_bf16(ah.v, bl, c[nsub]);  // hi*lo
        }
    }

    // agg and out alias (d_out): all reads above are consumed by WMMAs before
    // this barrier; only this block touches rows [mbase, mbase+16).
    __syncthreads();

#pragma unroll
    for (int nsub = 0; nsub < 4; ++nsub) {
        int n = nbase + (nsub << 4) + l16;
        float bv = bias[n];
#pragma unroll
        for (int r = 0; r < 8; ++r) {
            out[(size_t)(mbase + r + half * 8) * D_FEAT + n] = c[nsub][r] + bv;
        }
    }
}

// ---------------------------------------------------------------------------
extern "C" void kernel_launch(void* const* d_in, const int* in_sizes, int n_in,
                              void* d_out, int out_size, void* d_ws, size_t ws_size,
                              hipStream_t stream) {
    const float* x      = (const float*)d_in[0];
    const float* weight = (const float*)d_in[1];
    const float* bias   = (const float*)d_in[2];
    const int*   src    = (const int*)d_in[3];
    const int*   dst    = (const int*)d_in[4];
    float* out = (float*)d_out;

    const int n_edges = in_sizes[3];

    // Workspace: bf16 hi/lo permuted weights (512 KB each)
    unsigned short* w_hi = (unsigned short*)d_ws;
    unsigned short* w_lo = w_hi + (size_t)D_FEAT * D_FEAT;

    // Phase 1: zero agg (= d_out)
    const int n4 = N_NODES * D_FEAT / 4;                 // 6.4M float4
    zero_kernel<<<25000, 256, 0, stream>>>((float4*)out, n4);

    // Phase 2: split+permute weights
    wsplit_kernel<<<(D_FEAT * D_FEAT) / 256, 256, 0, stream>>>(weight, w_hi, w_lo);

    // Phase 3: gather + atomic scatter-add (1 wave per edge)
    scatter_kernel<<<(n_edges + 7) / 8, 256, 0, stream>>>(x, src, dst, out, n_edges);

    // Phase 4: WMMA GEMM, in-place agg -> out
    gemm_kernel<<<N_NODES / 16, 256, 0, stream>>>(out, w_hi, w_lo, bias, out);
}
